// QAOnlyHopMemoryModel_39273180955465
// MI455X (gfx1250) — matrix operand order repaired and checked by reading, not compiled, not run on hardware
//
#include <hip/hip_runtime.h>
#include <math.h>

// ---------------- model constants ----------------
constexpr int kB   = 32;     // batch
constexpr int kS   = 512;    // max seq
constexpr int kT   = 50;     // topic memory tokens
constexpr int kD   = 512;    // hidden
constexpr int kH   = 8;      // heads
constexpr int kDH  = 64;     // head dim
constexpr int kF   = 2048;   // ffn
constexpr int kNL  = 4;      // layers
constexpr int kVB  = 30000;  // bow vocab
constexpr int kL   = kS + kT;        // 562 real positions
constexpr int kLP  = 576;            // padded to 16 (36 tiles)
constexpr int kM   = kB * kLP;       // 18432 padded tokens (144 * 128)

// ---------------- CDNA5 async-to-LDS availability ----------------
#if defined(__has_builtin)
#  if __has_builtin(__builtin_amdgcn_global_load_async_to_lds_b128)
#    define HAVE_ASYNC_LDS 1
#  endif
#endif
#ifndef HAVE_ASYNC_LDS
#  define HAVE_ASYNC_LDS 0
#endif

typedef int v4i_t __attribute__((ext_vector_type(4)));
typedef __attribute__((address_space(1))) v4i_t gv4i;  // global int4
typedef __attribute__((address_space(3))) v4i_t lv4i;  // LDS int4

__device__ __forceinline__ void wait_async_all() {
#if HAVE_ASYNC_LDS
#  if __has_builtin(__builtin_amdgcn_s_wait_asynccnt)
  __builtin_amdgcn_s_wait_asynccnt(0);
#  else
  asm volatile("s_wait_asynccnt 0" ::: "memory");
#  endif
#endif
}

// ---------------- WMMA types / helpers ----------------
typedef __attribute__((ext_vector_type(16))) __bf16 v16bf;
typedef __attribute__((ext_vector_type(8)))  float  v8f;

__device__ __forceinline__ v8f wmma_bf16(v16bf a, v16bf b, v8f c) {
  // D = A(16x32 bf16) * B(32x16 bf16) + C(16x16 f32)
  return __builtin_amdgcn_wmma_f32_16x16x32_bf16(
      false, a, false, b, (short)0, c, false, false);
}

// A-fragment (16x32, row-major source with leading dim ld), contiguous K order (k = 16*hi + s), same as the B loaders:
// lane lr (0..15) holds row M=lr; lanes 0-15 hold K 0..15, lanes 16-31 hold K 16..31;
// 2 consecutive K packed per VGPR.
__device__ __forceinline__ v16bf load_a_frag(const __bf16* src, int ld) {
  int lane = threadIdx.x & 31;
  int lr = lane & 15, hi = lane >> 4;
  v16bf a;
#pragma unroll
  for (int j = 0; j < 8; ++j) {
    int k = hi * 16 + 2 * j;
    a[2 * j]     = src[lr * ld + k];
    a[2 * j + 1] = src[lr * ld + k + 1];
  }
  return a;
}

// B-fragment (32x16, row-major source with leading dim ld), contiguous K order (k = 16*hi + s), same as load_a_frag:
// lanes 0-15 hold K 0..15 (2 consecutive K per VGPR), lanes 16-31 hold K 16..31;
// N column = lane % 16.
__device__ __forceinline__ v16bf load_b_frag(const __bf16* src, int ld) {
  int lane = threadIdx.x & 31;
  int lr = lane & 15, hi = lane >> 4;
  v16bf b;
#pragma unroll
  for (int j = 0; j < 8; ++j) {
    int k = hi * 16 + 2 * j;
    b[2 * j]     = src[k * ld + lr];
    b[2 * j + 1] = src[(k + 1) * ld + lr];
  }
  return b;
}

// B-fragment for K^T: B[k][n] = Kmat[n*ld + cbase + k]  (reads K rows contiguously)
__device__ __forceinline__ v16bf load_bT_frag(const __bf16* kmat, int ld, int cbase) {
  int lane = threadIdx.x & 31;
  int lr = lane & 15, hi = lane >> 4;
  v16bf b;
#pragma unroll
  for (int j = 0; j < 8; ++j) {
    int c = cbase + hi * 16 + 2 * j;
    b[2 * j]     = kmat[lr * ld + c];
    b[2 * j + 1] = kmat[lr * ld + c + 1];
  }
  return b;
}

// ---------------- small kernels ----------------
__global__ void f32_to_bf16_k(const float* __restrict__ src, __bf16* __restrict__ dst,
                              size_t n) {
  size_t i = (size_t)blockIdx.x * 256 + threadIdx.x;
  if (i < n) dst[i] = (__bf16)src[i];
}

__global__ __launch_bounds__(256) void topic_softmax_k(const float* __restrict__ Wd,
                                                       const float* __restrict__ bd,
                                                       float* __restrict__ outp) {
  __shared__ float red[256];
  int t = blockIdx.x, tid = threadIdx.x;
  const float* rowp = Wd + (size_t)t * kVB;
  float* orow = outp + (size_t)t * kVB;
  float mx = -1e30f;
  for (int i = tid; i < kVB; i += 256) mx = fmaxf(mx, rowp[i] + bd[i]);
  red[tid] = mx; __syncthreads();
  for (int s = 128; s > 0; s >>= 1) { if (tid < s) red[tid] = fmaxf(red[tid], red[tid + s]); __syncthreads(); }
  float rmax = red[0]; __syncthreads();
  float sum = 0.f;
  for (int i = tid; i < kVB; i += 256) { float e = __expf(rowp[i] + bd[i] - rmax); orow[i] = e; sum += e; }
  red[tid] = sum; __syncthreads();
  for (int s = 128; s > 0; s >>= 1) { if (tid < s) red[tid] += red[tid + s]; __syncthreads(); }
  float inv = 1.f / red[0];
  for (int i = tid; i < kVB; i += 256) orow[i] *= inv;
}

__global__ __launch_bounds__(128) void mem_mlp1_k(const float* __restrict__ wd,
                                                  const float* __restrict__ W1,
                                                  const float* __restrict__ b1,
                                                  float* __restrict__ hm) {
  int t = blockIdx.y;
  int j = blockIdx.x * 128 + threadIdx.x;   // 0..1023
  const float* wr = wd + (size_t)t * kVB;
  float s = 0.f;
  for (int k2 = 0; k2 < kVB; ++k2) s = fmaf(wr[k2], W1[(size_t)k2 * (2 * kD) + j], s);
  s += b1[j];
  hm[t * (2 * kD) + j] = fmaxf(s, 0.f);
}

__global__ __launch_bounds__(256) void mem_mlp2_ln_k(const float* __restrict__ hm,
                                                     const float* __restrict__ W2,
                                                     const float* __restrict__ b2,
                                                     const float* __restrict__ g,
                                                     const float* __restrict__ bb,
                                                     float* __restrict__ topic) {
  __shared__ float red[256];
  int t = blockIdx.x, tid = threadIdx.x;
  const float* hr = hm + t * (2 * kD);
  float a0 = 0.f, a1 = 0.f;
  for (int k2 = 0; k2 < 2 * kD; ++k2) {
    float hv = hr[k2];
    a0 = fmaf(hv, W2[(size_t)k2 * kD + tid], a0);
    a1 = fmaf(hv, W2[(size_t)k2 * kD + tid + 256], a1);
  }
  a0 += b2[tid]; a1 += b2[tid + 256];
  red[tid] = a0 + a1; __syncthreads();
  for (int s = 128; s > 0; s >>= 1) { if (tid < s) red[tid] += red[tid + s]; __syncthreads(); }
  float mean = red[0] * (1.f / kD); __syncthreads();
  float d0 = a0 - mean, d1 = a1 - mean;
  red[tid] = d0 * d0 + d1 * d1; __syncthreads();
  for (int s = 128; s > 0; s >>= 1) { if (tid < s) red[tid] += red[tid + s]; __syncthreads(); }
  float rstd = rsqrtf(red[0] * (1.f / kD) + 1e-12f);
  topic[t * kD + tid]       = d0 * rstd * g[tid]       + bb[tid];
  topic[t * kD + tid + 256] = d1 * rstd * g[tid + 256] + bb[tid + 256];
}

__global__ __launch_bounds__(128) void build_hidden_k(const int* __restrict__ ids,
                                                      const int* __restrict__ lens,
                                                      const float* __restrict__ embed,
                                                      const float* __restrict__ topic,
                                                      float* __restrict__ hidden,
                                                      __bf16* __restrict__ hbf,
                                                      float* __restrict__ biasb) {
  int b = blockIdx.y, j = blockIdx.x, tid = threadIdx.x;
  int len = lens[b]; len = min(max(len, 1), kS);
  const float* src = nullptr;
  bool zero = false;
  if (j < len)              src = embed + (size_t)ids[b * kS + j] * kD;
  else if (j < len + kT)    src = topic + (size_t)(j - len) * kD;
  else if (j < kL)          src = embed + (size_t)ids[b * kS + (j - kT)] * kD;
  else                      zero = true;
  size_t off = ((size_t)b * kLP + j) * kD;
#pragma unroll
  for (int i = 0; i < kD / 128; ++i) {
    int d = tid + i * 128;
    float v = zero ? 0.f : src[d];
    hidden[off + d] = v;
    hbf[off + d] = (__bf16)v;
  }
  if (tid == 0) biasb[b * kLP + j] = (j < len + kT) ? 0.f : -10000.f;
}

__global__ __launch_bounds__(256) void ln_residual_k(float* __restrict__ hidden,
                                                     const float* __restrict__ add,
                                                     const float* __restrict__ g,
                                                     const float* __restrict__ bta,
                                                     __bf16* __restrict__ hbf) {
  __shared__ float red[256];
  int row = blockIdx.x, tid = threadIdx.x;
  size_t off = (size_t)row * kD;
  float x0 = hidden[off + tid]       + add[off + tid];
  float x1 = hidden[off + tid + 256] + add[off + tid + 256];
  red[tid] = x0 + x1; __syncthreads();
  for (int s = 128; s > 0; s >>= 1) { if (tid < s) red[tid] += red[tid + s]; __syncthreads(); }
  float mean = red[0] * (1.f / kD); __syncthreads();
  float d0 = x0 - mean, d1 = x1 - mean;
  red[tid] = d0 * d0 + d1 * d1; __syncthreads();
  for (int s = 128; s > 0; s >>= 1) { if (tid < s) red[tid] += red[tid + s]; __syncthreads(); }
  float rstd = rsqrtf(red[0] * (1.f / kD) + 1e-12f);
  float y0 = d0 * rstd * g[tid]       + bta[tid];
  float y1 = d1 * rstd * g[tid + 256] + bta[tid + 256];
  hidden[off + tid] = y0;       hbf[off + tid] = (__bf16)y0;
  hidden[off + tid + 256] = y1; hbf[off + tid + 256] = (__bf16)y1;
}

__global__ void copy_out_k(const float* __restrict__ hidden, float* __restrict__ out) {
  size_t i = (size_t)blockIdx.x * 256 + threadIdx.x;
  if (i >= (size_t)kB * kL * kD) return;
  int d = (int)(i % kD);
  size_t r = i / kD;
  int j = (int)(r % kL);
  int b = (int)(r / kL);
  out[i] = hidden[((size_t)b * kLP + j) * kD + d];
}

// ---------------- bf16 WMMA GEMM (double-buffered, async-to-LDS staging) ----------------
// C[M,N] = epilogue(A[M,K](bf16,row-major) * W[K,N](bf16,row-major) + bias[N])
// Block: 256 threads = 8 waves (4 in M x 2 in N). Tile 128x64x32; each wave 32x32.
constexpr int TBM = 128, TBN = 64, TBK = 32;
constexpr int TLDA = 40, TLDB = 72;   // bf16 elements, padded against bank conflicts

__device__ __forceinline__ void stage_tile(const __bf16* __restrict__ A,
                                           const __bf16* __restrict__ W,
                                           int K, int N, int rowBase, int colBase, int kt,
                                           __bf16* as, __bf16* bs,
                                           int arow, int acol, int brow, int bcol) {
#if HAVE_ASYNC_LDS
  // ASYNCcnt-tracked copy engine path: 16B per lane, no VGPR staging.
  __builtin_amdgcn_global_load_async_to_lds_b128(
      (gv4i*)(__bf16*)&A[(size_t)(rowBase + arow) * K + kt + acol],
      (lv4i*)&as[arow * TLDA + acol], 0, 0);
  __builtin_amdgcn_global_load_async_to_lds_b128(
      (gv4i*)(__bf16*)&A[(size_t)(rowBase + arow + 64) * K + kt + acol],
      (lv4i*)&as[(arow + 64) * TLDA + acol], 0, 0);
  __builtin_amdgcn_global_load_async_to_lds_b128(
      (gv4i*)(__bf16*)&W[(size_t)(kt + brow) * N + colBase + bcol],
      (lv4i*)&bs[brow * TLDB + bcol], 0, 0);
#else
  *reinterpret_cast<uint4*>(&as[arow * TLDA + acol]) =
      *reinterpret_cast<const uint4*>(&A[(size_t)(rowBase + arow) * K + kt + acol]);
  *reinterpret_cast<uint4*>(&as[(arow + 64) * TLDA + acol]) =
      *reinterpret_cast<const uint4*>(&A[(size_t)(rowBase + arow + 64) * K + kt + acol]);
  *reinterpret_cast<uint4*>(&bs[brow * TLDB + bcol]) =
      *reinterpret_cast<const uint4*>(&W[(size_t)(kt + brow) * N + colBase + bcol]);
#endif
}

// EPI: 0 = f32 out; 1 = QKV scatter bf16 (scaled); 2 = GELU -> bf16
template <int EPI>
__global__ __launch_bounds__(256) void gemm_bf16_k(const __bf16* __restrict__ A,
                                                   const __bf16* __restrict__ W,
                                                   const float* __restrict__ bias,
                                                   void* __restrict__ out,
                                                   int M, int N, int K, float scale) {
  __shared__ alignas(16) __bf16 As[2][TBM * TLDA];
  __shared__ alignas(16) __bf16 Bs[2][TBK * TLDB];
  int tid = threadIdx.x;
  int lane = tid & 31, wave = tid >> 5;
  int wm = wave >> 1, wn = wave & 1;
  int lr = lane & 15, hi = lane >> 4;
  int rowBase = blockIdx.y * TBM;
  int colBase = blockIdx.x * TBN;

  v8f zero8;
#pragma unroll
  for (int i = 0; i < 8; ++i) zero8[i] = 0.f;
  v8f acc[4] = {zero8, zero8, zero8, zero8};

  int arow = tid >> 2;        // 0..63
  int acol = (tid & 3) * 8;   // 0,8,16,24
  int brow = tid >> 3;        // 0..31
  int bcol = (tid & 7) * 8;   // 0..56

  // prologue: stage tile 0 into buffer 0
  stage_tile(A, W, K, N, rowBase, colBase, 0, As[0], Bs[0], arow, acol, brow, bcol);
  wait_async_all();
  __syncthreads();

  int cur = 0;
  for (int kt = 0; kt < K; kt += TBK) {
    // stage next tile into the other buffer while computing on this one
    if (kt + TBK < K)
      stage_tile(A, W, K, N, rowBase, colBase, kt + TBK,
                 As[cur ^ 1], Bs[cur ^ 1], arow, acol, brow, bcol);

    v16bf a0 = load_a_frag(&As[cur][(wm * 32) * TLDA], TLDA);
    v16bf a1 = load_a_frag(&As[cur][(wm * 32 + 16) * TLDA], TLDA);
    v16bf b0 = load_b_frag(&Bs[cur][wn * 32], TLDB);
    v16bf b1 = load_b_frag(&Bs[cur][wn * 32 + 16], TLDB);
    acc[0] = wmma_bf16(a0, b0, acc[0]);
    acc[1] = wmma_bf16(a0, b1, acc[1]);
    acc[2] = wmma_bf16(a1, b0, acc[2]);
    acc[3] = wmma_bf16(a1, b1, acc[3]);

    wait_async_all();     // ensure our async fills of the next buffer landed
    __syncthreads();      // and that every wave is done reading/writing
    cur ^= 1;
  }

#pragma unroll
  for (int f = 0; f < 4; ++f) {
    int fm = f >> 1, fn = f & 1;
#pragma unroll
    for (int r = 0; r < 8; ++r) {
      int row = rowBase + wm * 32 + fm * 16 + r + hi * 8;
      int col = colBase + wn * 32 + fn * 16 + lr;
      float v = acc[f][r] + bias[col];
      if (EPI == 0) {
        ((float*)out)[(size_t)row * N + col] = v;
      } else if (EPI == 1) {
        int bidx = row / kLP, l = row % kLP;
        int h = col >> 6, d = col & 63;
        ((__bf16*)out)[(((size_t)bidx * kH + h) * kLP + l) * kDH + d] = (__bf16)(v * scale);
      } else {
        float gl = 0.5f * v * (1.f + erff(v * 0.70710678118f));
        ((__bf16*)out)[(size_t)row * N + col] = (__bf16)gl;
      }
    }
  }
}

// ---------------- flash-style WMMA attention ----------------
// grid (kLP/64, kH, kB), block 128 (4 waves, one 16-row q-tile each)
__global__ __launch_bounds__(128) void attention_k(const __bf16* __restrict__ Q,
                                                   const __bf16* __restrict__ Kb,
                                                   const __bf16* __restrict__ Vb,
                                                   const float* __restrict__ biasbuf,
                                                   __bf16* __restrict__ ctx) {
  __shared__ alignas(16) __bf16 pstage[4][16 * 40];
  int tid = threadIdx.x, lane = tid & 31, wave = tid >> 5;
  int lr = lane & 15, hi = lane >> 4;
  int b = blockIdx.z, h = blockIdx.y;
  int q0 = blockIdx.x * 64 + wave * 16;
  size_t base = ((size_t)b * kH + h) * kLP * kDH;
  const __bf16* qp = Q + base + (size_t)q0 * kDH;
  const __bf16* kp = Kb + base;
  const __bf16* vp = Vb + base;
  const float* bp = biasbuf + b * kLP;

  v16bf aq0 = load_a_frag(qp, kDH);        // Q cols 0..31 (scale already folded in)
  v16bf aq1 = load_a_frag(qp + 32, kDH);   // Q cols 32..63

  v8f zero8;
#pragma unroll
  for (int i = 0; i < 8; ++i) zero8[i] = 0.f;
  v8f acc[4] = {zero8, zero8, zero8, zero8};
  float m[8], l[8];
#pragma unroll
  for (int r = 0; r < 8; ++r) { m[r] = -1e30f; l[r] = 0.f; }

  __bf16* ps = pstage[wave];

  for (int kb0 = 0; kb0 < kLP; kb0 += 32) {
    if (kb0 + 32 < kLP) {   // pull next K/V chunk toward us (global_prefetch_b8)
      __builtin_prefetch(kp + (size_t)(kb0 + 32 + (tid & 31)) * kDH, 0, 3);
      __builtin_prefetch(vp + (size_t)(kb0 + 32 + (tid & 31)) * kDH, 0, 3);
    }
    v8f s0 = zero8, s1 = zero8;
    {
      v16bf bk = load_bT_frag(kp + (size_t)kb0 * kDH, kDH, 0);
      s0 = wmma_bf16(aq0, bk, s0);
      bk = load_bT_frag(kp + (size_t)kb0 * kDH, kDH, 32);
      s0 = wmma_bf16(aq1, bk, s0);
      bk = load_bT_frag(kp + (size_t)(kb0 + 16) * kDH, kDH, 0);
      s1 = wmma_bf16(aq0, bk, s1);
      bk = load_bT_frag(kp + (size_t)(kb0 + 16) * kDH, kDH, 32);
      s1 = wmma_bf16(aq1, bk, s1);
    }
    float bias0 = bp[kb0 + lr], bias1 = bp[kb0 + 16 + lr];
#pragma unroll
    for (int r = 0; r < 8; ++r) {
      float x0 = s0[r] + bias0, x1 = s1[r] + bias1;
      float rm = fmaxf(x0, x1);
      rm = fmaxf(rm, __shfl_xor(rm, 1, 32));
      rm = fmaxf(rm, __shfl_xor(rm, 2, 32));
      rm = fmaxf(rm, __shfl_xor(rm, 4, 32));
      rm = fmaxf(rm, __shfl_xor(rm, 8, 32));
      float mn = fmaxf(m[r], rm);
      float corr = __expf(m[r] - mn);
      float e0 = __expf(x0 - mn), e1 = __expf(x1 - mn);
      float rs = e0 + e1;
      rs += __shfl_xor(rs, 1, 32);
      rs += __shfl_xor(rs, 2, 32);
      rs += __shfl_xor(rs, 4, 32);
      rs += __shfl_xor(rs, 8, 32);
      l[r] = l[r] * corr + rs;
      m[r] = mn;
      acc[0][r] *= corr; acc[1][r] *= corr; acc[2][r] *= corr; acc[3][r] *= corr;
      ps[(r + hi * 8) * 40 + lr]      = (__bf16)e0;
      ps[(r + hi * 8) * 40 + 16 + lr] = (__bf16)e1;
    }
    // wave-local LDS RAW: make P stores visible before fragment reads
    asm volatile("s_wait_dscnt 0" ::: "memory");
    v16bf ap = load_a_frag(ps, 40);
#pragma unroll
    for (int nf = 0; nf < 4; ++nf) {
      v16bf bv = load_b_frag(vp + (size_t)kb0 * kDH + nf * 16, kDH);
      acc[nf] = wmma_bf16(ap, bv, acc[nf]);
    }
  }
#pragma unroll
  for (int r = 0; r < 8; ++r) {
    float inv = 1.f / l[r];
    size_t row = (size_t)b * kLP + q0 + r + hi * 8;
#pragma unroll
    for (int nf = 0; nf < 4; ++nf)
      ctx[row * kD + h * kDH + nf * 16 + lr] = (__bf16)(acc[nf][r] * inv);
  }
}

// ---------------- launcher ----------------
extern "C" void kernel_launch(void* const* d_in, const int* in_sizes, int n_in,
                              void* d_out, int out_size, void* d_ws, size_t ws_size,
                              hipStream_t stream) {
  const int*   ids   = (const int*)d_in[0];
  const int*   lens  = (const int*)d_in[1];
  const float* embed = (const float*)d_in[2];
  const float* W_dec = (const float*)d_in[3];
  const float* b_dec = (const float*)d_in[4];
  const float* mW1   = (const float*)d_in[5];
  const float* mb1   = (const float*)d_in[6];
  const float* mW2   = (const float*)d_in[7];
  const float* mb2   = (const float*)d_in[8];
  const float* mlg   = (const float*)d_in[9];
  const float* mlb   = (const float*)d_in[10];
  const float* Wq    = (const float*)d_in[11];
  const float* bq    = (const float*)d_in[12];
  const float* Wk    = (const float*)d_in[13];
  const float* bk    = (const float*)d_in[14];
  const float* Wv    = (const float*)d_in[15];
  const float* bv    = (const float*)d_in[16];
  const float* Wo    = (const float*)d_in[17];
  const float* bo    = (const float*)d_in[18];
  const float* ln1g  = (const float*)d_in[19];
  const float* ln1b  = (const float*)d_in[20];
  const float* Wi    = (const float*)d_in[21];
  const float* bi    = (const float*)d_in[22];
  const float* Wf    = (const float*)d_in[23];
  const float* bFv   = (const float*)d_in[24];
  const float* ln2g  = (const float*)d_in[25];
  const float* ln2b  = (const float*)d_in[26];
  (void)in_sizes; (void)n_in; (void)out_size; (void)ws_size;

  char* p = (char*)d_ws;
  auto carve = [&](size_t bytes) -> char* {
    char* r = p;
    p += (bytes + 255) & ~(size_t)255;
    return r;
  };
  float*  ws_wd     = (float*)carve((size_t)kT * kVB * 4);
  float*  ws_hmem   = (float*)carve((size_t)kT * 2 * kD * 4);
  float*  ws_topic  = (float*)carve((size_t)kT * kD * 4);
  float*  ws_bias   = (float*)carve((size_t)kB * kLP * 4);
  float*  ws_hidden = (float*)carve((size_t)kM * kD * 4);
  __bf16* ws_hbf    = (__bf16*)carve((size_t)kM * kD * 2);
  float*  ws_tmp    = (float*)carve((size_t)kM * kD * 4);
  __bf16* ws_ff     = (__bf16*)carve((size_t)kM * kF * 2);
  __bf16* ws_q      = (__bf16*)carve((size_t)kM * kD * 2);
  __bf16* ws_k      = (__bf16*)carve((size_t)kM * kD * 2);
  __bf16* ws_v      = (__bf16*)carve((size_t)kM * kD * 2);
  __bf16* ws_ctx    = (__bf16*)carve((size_t)kM * kD * 2);
  __bf16* ws_wq     = (__bf16*)carve((size_t)kNL * kD * kD * 2);
  __bf16* ws_wk     = (__bf16*)carve((size_t)kNL * kD * kD * 2);
  __bf16* ws_wv     = (__bf16*)carve((size_t)kNL * kD * kD * 2);
  __bf16* ws_wo     = (__bf16*)carve((size_t)kNL * kD * kD * 2);
  __bf16* ws_wi     = (__bf16*)carve((size_t)kNL * kD * kF * 2);
  __bf16* ws_wf     = (__bf16*)carve((size_t)kNL * kF * kD * 2);

  // ---- topic memory ----
  topic_softmax_k<<<kT, 256, 0, stream>>>(W_dec, b_dec, ws_wd);
  mem_mlp1_k<<<dim3(2 * kD / 128, kT), 128, 0, stream>>>(ws_wd, mW1, mb1, ws_hmem);
  mem_mlp2_ln_k<<<kT, 256, 0, stream>>>(ws_hmem, mW2, mb2, mlg, mlb, ws_topic);

  // ---- weight conversion to bf16 ----
  size_t nqk = (size_t)kNL * kD * kD;
  size_t nif = (size_t)kNL * kD * kF;
  int gq = (int)((nqk + 255) / 256), gi = (int)((nif + 255) / 256);
  f32_to_bf16_k<<<gq, 256, 0, stream>>>(Wq, ws_wq, nqk);
  f32_to_bf16_k<<<gq, 256, 0, stream>>>(Wk, ws_wk, nqk);
  f32_to_bf16_k<<<gq, 256, 0, stream>>>(Wv, ws_wv, nqk);
  f32_to_bf16_k<<<gq, 256, 0, stream>>>(Wo, ws_wo, nqk);
  f32_to_bf16_k<<<gi, 256, 0, stream>>>(Wi, ws_wi, nif);
  f32_to_bf16_k<<<gi, 256, 0, stream>>>(Wf, ws_wf, nif);

  // ---- build padded hidden + attention bias ----
  build_hidden_k<<<dim3(kLP, kB), 128, 0, stream>>>(ids, lens, embed, ws_topic,
                                                    ws_hidden, ws_hbf, ws_bias);

  dim3 gD(kD / TBN, kM / TBM);   // (8, 144)
  dim3 gF(kF / TBN, kM / TBM);   // (32, 144)
  const float qscale = 0.125f;   // 1/sqrt(64)

  for (int L = 0; L < kNL; ++L) {
    const __bf16* wq = ws_wq + (size_t)L * kD * kD;
    const __bf16* wk = ws_wk + (size_t)L * kD * kD;
    const __bf16* wv = ws_wv + (size_t)L * kD * kD;
    const __bf16* wo = ws_wo + (size_t)L * kD * kD;
    const __bf16* wi = ws_wi + (size_t)L * kD * kF;
    const __bf16* wf = ws_wf + (size_t)L * kF * kD;

    gemm_bf16_k<1><<<gD, 256, 0, stream>>>(ws_hbf, wq, bq + L * kD, ws_q, kM, kD, kD, qscale);
    gemm_bf16_k<1><<<gD, 256, 0, stream>>>(ws_hbf, wk, bk + L * kD, ws_k, kM, kD, kD, 1.f);
    gemm_bf16_k<1><<<gD, 256, 0, stream>>>(ws_hbf, wv, bv + L * kD, ws_v, kM, kD, kD, 1.f);

    attention_k<<<dim3(kLP / 64, kH, kB), 128, 0, stream>>>(ws_q, ws_k, ws_v, ws_bias, ws_ctx);

    gemm_bf16_k<0><<<gD, 256, 0, stream>>>(ws_ctx, wo, bo + L * kD, ws_tmp, kM, kD, kD, 1.f);
    ln_residual_k<<<kM, 256, 0, stream>>>(ws_hidden, ws_tmp, ln1g + L * kD, ln1b + L * kD, ws_hbf);

    gemm_bf16_k<2><<<gF, 256, 0, stream>>>(ws_hbf, wi, bi + L * kF, ws_ff, kM, kF, kD, 1.f);
    gemm_bf16_k<0><<<gD, 256, 0, stream>>>(ws_ff, wf, bFv + L * kD, ws_tmp, kM, kD, kF, 1.f);
    ln_residual_k<<<kM, 256, 0, stream>>>(ws_hidden, ws_tmp, ln2g + L * kD, ln2b + L * kD, ws_hbf);
  }

  size_t nout = (size_t)kB * kL * kD;
  copy_out_k<<<(int)((nout + 255) / 256), 256, 0, stream>>>(ws_hidden, (float*)d_out);
}
